// LearnablePermutation_53678501265934
// MI455X (gfx1250) — matrix operand order repaired and checked by reading, not compiled, not run on hardware
//
#include <hip/hip_runtime.h>

typedef _Float16 v16h __attribute__((ext_vector_type(16)));
typedef float    v8f  __attribute__((ext_vector_type(8)));
typedef unsigned int u32x4 __attribute__((ext_vector_type(4)));
typedef int      i32x4 __attribute__((ext_vector_type(4)));
typedef int      i32x8 __attribute__((ext_vector_type(8)));

#define SINK_ITERS 50
#define RS 66                      // padded row stride (floats): banks (2r+j)%64 -> conflict-free
#define E_OFF   (64 * RS)          // packed (hi,lo) f16 of 2^u, same padded indexing
#define FR_OFF  (E_OFF + 64 * RS)  // 64 packed (hi,lo) row factors 2^(-lr)
#define LR_OFF  (FR_OFF + 64)
#define LC_OFF  (LR_OFF + 64)
#define NFLOATS (LC_OFF + 64)      // 8640 floats = 34560 B static LDS

#define TEN_LOG2E 14.426950408889634f  // 10 / ln(2): temp scale folded into base-2 domain
#define LOG2E 1.4426950408889634f

#define LDS_FENCE() asm volatile("s_wait_dscnt 0" ::: "memory")

__device__ __forceinline__ float fexp2(float x) { return __builtin_amdgcn_exp2f(x); }
__device__ __forceinline__ float flog2(float x) { return __builtin_amdgcn_logf(x); }
__device__ __forceinline__ _Float16 lo16(unsigned w) {
    return __builtin_bit_cast(_Float16, (unsigned short)(w & 0xFFFFu));
}
__device__ __forceinline__ _Float16 hi16(unsigned w) {
    return __builtin_bit_cast(_Float16, (unsigned short)(w >> 16));
}
__device__ __forceinline__ unsigned pack2(_Float16 h, _Float16 l) {
    return (unsigned)__builtin_bit_cast(unsigned short, h) |
           ((unsigned)__builtin_bit_cast(unsigned short, l) << 16);
}

// ---- Tensor Data Mover: 64x64 f32 tile between global and LDS.
// On load, pad_interval=64 dwords / pad_amount=2 dwords makes the TDM itself
// produce the stride-66 bank-conflict-free LDS layout.
__device__ __forceinline__ void tdm_tile_2d(int is_store, unsigned lds_byte,
                                            unsigned long long ga, int pad)
{
    u32x4 g0;
    g0[0] = 1u;                                                 // count=1, user descriptor
    g0[1] = lds_byte;                                           // lds_addr
    g0[2] = (unsigned)ga;                                       // global_addr[31:0]
    g0[3] = (unsigned)((ga >> 32) & 0x01FFFFFFu) | (2u << 30);  // addr[56:32] | type=2
    i32x8 g1;
    g1[0] = (2 << 16) | (pad ? ((1 << 20) | (5 << 22) | (1 << 25)) : 0); // 4B elems [+pad]
    g1[1] = (64 << 16);   // tensor_dim0 = 64
    g1[2] = (64 << 16);   // tensor_dim1 = 64
    g1[3] = (64 << 16);   // tile_dim0 = 64
    g1[4] = 64;           // tile_dim1 = 64 (tile_dim2 = 0)
    g1[5] = 64;           // tensor_dim0_stride = 64
    g1[6] = 0;
    g1[7] = 0;
    i32x4 gz = {0, 0, 0, 0};
#if __has_include(<hip/amd_detail/amd_gfx1250_TDM.h>)
    i32x8 gz8 = {0, 0, 0, 0, 0, 0, 0, 0};
    if (is_store) __builtin_amdgcn_tensor_store_from_lds(g0, g1, gz, gz, gz8, 0);
    else          __builtin_amdgcn_tensor_load_to_lds  (g0, g1, gz, gz, gz8, 0);
#else
    if (is_store) __builtin_amdgcn_tensor_store_from_lds(g0, g1, gz, gz, 0);
    else          __builtin_amdgcn_tensor_load_to_lds  (g0, g1, gz, gz, 0);
#endif
}

__global__ __launch_bounds__(32) void
sinkhorn_wmma_kernel(const float* __restrict__ gamma,
                     const float* __restrict__ noise,
                     float* __restrict__ out)
{
    __shared__ float lds[NFLOATS];
    const int lane = threadIdx.x & 31;
    const int b    = blockIdx.x;

    float*    la  = lds;                          // u = la * log2e, 64x64 stride RS
    float*    spb = lds + E_OFF;                  // staging: softplus terms / output tile
    unsigned* Eb  = (unsigned*)(lds + E_OFF);     // packed (hi,lo) f16 of 2^u
    unsigned* fr  = (unsigned*)(lds + FR_OFF);    // packed (hi,lo) f16 of 2^(-lr)
    float*    lr2 = lds + LR_OFF;
    float*    lc2 = lds + LC_OFF;

    // ---- TDM: DMA the noise tile into la region (padded stride-66) ...
    tdm_tile_2d(0, 0u, (unsigned long long)(noise + (size_t)b * 4096), 1);

    // ... overlapped with gamma softplus in base-2 domain:
    // 10*log2e * softplus(-g) = 10*log2(1 + 2^(-|g|*log2e)) + (g<0 ? -g*10*log2e : 0)
    #pragma unroll 4
    for (int t = 0; t < 128; ++t) {
        int e = t * 32 + lane;
        float g = gamma[e];
        float s = 10.0f * flog2(1.0f + fexp2(-fabsf(g) * LOG2E));
        spb[e] = s + (g < 0.0f ? -g * TEN_LOG2E : 0.0f);
    }
    __builtin_amdgcn_s_wait_tensorcnt(0);
    LDS_FENCE();

    // combine: u = softplus_term + noise * 10*log2e
    #pragma unroll 4
    for (int t = 0; t < 128; ++t) {
        int e = t * 32 + lane;
        int a = (e >> 6) * RS + (e & 63);
        la[a] = spb[e] + la[a] * TEN_LOG2E;
    }
    LDS_FENCE();

    // ---- one-time per-row max subtraction (invariant under row normalization);
    // afterwards all entries stay <= 0, so plain sum-of-exp2 is safe forever.
    #pragma unroll
    for (int rr = 0; rr < 2; ++rr) {
        int r = lane + rr * 32;
        float mx = -3.0e38f;
        #pragma unroll
        for (int j = 0; j < 64; j += 2) {
            float2 p = *(const float2*)&la[r * RS + j];
            mx = fmaxf(mx, fmaxf(p.x, p.y));
        }
        #pragma unroll
        for (int j = 0; j < 64; j += 2) {
            float2* p = (float2*)&la[r * RS + j];
            float2 v = *p; v.x -= mx; v.y -= mx; *p = v;
        }
    }
    LDS_FENCE();

    v16h ones;
    #pragma unroll
    for (int s = 0; s < 16; ++s) ones[s] = (_Float16)1.0f;

    const int rm   = lane & 15;
    const int half = lane >> 4;

    #pragma unroll 1
    for (int it = 0; it < SINK_ITERS; ++it) {
        // ===== row pass: lr2[r] = log2(sum_c 2^u[r][c]) via E x ones; cache E packed =====
        #pragma unroll
        for (int blk = 0; blk < 4; ++blk) {
            int r = blk * 16 + rm;
            v8f acc = {0.f, 0.f, 0.f, 0.f, 0.f, 0.f, 0.f, 0.f};
            #pragma unroll
            for (int kc = 0; kc < 2; ++kc) {
                v16h ahi, alo;  // A 16x32 f16 layout: lanes0-15 K={0..7,16..23}, lanes16-31 +8
                #pragma unroll
                for (int v = 0; v < 8; ++v) {
                    int k0 = ((v < 4) ? 2 * v : 8 + 2 * v) + 8 * half + kc * 32;
                    float2 p = *(const float2*)&la[r * RS + k0];
                    float e0 = fexp2(p.x), e1 = fexp2(p.y);
                    _Float16 h0 = (_Float16)e0, h1 = (_Float16)e1;
                    _Float16 l0 = (_Float16)(e0 - (float)h0), l1 = (_Float16)(e1 - (float)h1);
                    ahi[2 * v] = h0; ahi[2 * v + 1] = h1;
                    alo[2 * v] = l0; alo[2 * v + 1] = l1;
                    uint2 st; st.x = pack2(h0, l0); st.y = pack2(h1, l1);
                    *(uint2*)&Eb[r * RS + k0] = st;          // cache E for the col pass
                }
                acc = __builtin_amdgcn_wmma_f32_16x16x32_f16(false, ahi, false, ones,
                                                             (short)0, acc, false, false);
                acc = __builtin_amdgcn_wmma_f32_16x16x32_f16(false, alo, false, ones,
                                                             (short)0, acc, false, false);
            }
            if (rm == 0) {   // D: VGPR g, lane l -> row M = g + 8*(l>>4)
                #pragma unroll
                for (int g = 0; g < 8; ++g)
                    lr2[blk * 16 + half * 8 + g] = flog2(acc[g]);
            }
        }
        LDS_FENCE();

        // ===== per-row factors 2^(-lr), split hi/lo (64 exps/matrix vs 4096 saved) =====
        #pragma unroll
        for (int rr = 0; rr < 2; ++rr) {
            int r = lane + rr * 32;
            float f = fexp2(-lr2[r]);
            _Float16 fh = (_Float16)f;
            fr[r] = pack2(fh, (_Float16)(f - (float)fh));
        }
        LDS_FENCE();

        // ===== col pass: lc2[n] = log2(sum_r f_r * E[r][n]); factors ride in A operand =====
        #pragma unroll
        for (int blk = 0; blk < 4; ++blk) {
            int n = blk * 16 + rm;
            v8f acc = {0.f, 0.f, 0.f, 0.f, 0.f, 0.f, 0.f, 0.f};
            #pragma unroll
            for (int kc = 0; kc < 2; ++kc) {
                v16h fhi, flo, bhi, blo;
                #pragma unroll
                for (int v = 0; v < 8; ++v) {
                    // A[m][k] = f_k broadcast over m; K map matches B: lanes0-15 K=0..15, lanes16-31 K=16..31
                    int ka = 2 * v + 16 * half + kc * 32;
                    unsigned wf0 = fr[ka], wf1 = fr[ka + 1];
                    fhi[2 * v] = lo16(wf0); flo[2 * v] = hi16(wf0);
                    fhi[2 * v + 1] = lo16(wf1); flo[2 * v + 1] = hi16(wf1);
                    // B K order used here (contiguous, as A): lanes0-15 K=0..15, lanes16-31 K=16..31
                    int kb = 2 * v + 16 * half + kc * 32;
                    unsigned wb0 = Eb[kb * RS + n], wb1 = Eb[(kb + 1) * RS + n];
                    bhi[2 * v] = lo16(wb0); blo[2 * v] = hi16(wb0);
                    bhi[2 * v + 1] = lo16(wb1); blo[2 * v + 1] = hi16(wb1);
                }
                // (fhi+flo)x(Ehi+Elo), dropping flo x Elo (~2^-22 relative)
                acc = __builtin_amdgcn_wmma_f32_16x16x32_f16(false, fhi, false, bhi,
                                                             (short)0, acc, false, false);
                acc = __builtin_amdgcn_wmma_f32_16x16x32_f16(false, fhi, false, blo,
                                                             (short)0, acc, false, false);
                acc = __builtin_amdgcn_wmma_f32_16x16x32_f16(false, flo, false, bhi,
                                                             (short)0, acc, false, false);
            }
            if (half == 0) lc2[n] = flog2(acc[0]);
        }
        LDS_FENCE();

        // ===== apply both normalizations in log2 domain: u -= lr2[row] + lc2[col] =====
        #pragma unroll
        for (int rr = 0; rr < 2; ++rr) {
            int r = lane + rr * 32;
            float lrv = lr2[r];
            #pragma unroll
            for (int j = 0; j < 64; j += 2) {
                float2* p = (float2*)&la[r * RS + j];
                float2 v = *p;
                float2 c = *(const float2*)&lc2[j];
                v.x -= lrv + c.x;
                v.y -= lrv + c.y;
                *p = v;
            }
        }
        LDS_FENCE();
    }

    // ---- epilogue: stage 2^u contiguously in LDS, then TDM store to global
    #pragma unroll 4
    for (int t = 0; t < 128; ++t) {
        int e = t * 32 + lane;
        spb[e] = fexp2(la[(e >> 6) * RS + (e & 63)]);
    }
    LDS_FENCE();
    tdm_tile_2d(1, (unsigned)(E_OFF * 4), (unsigned long long)(out + (size_t)b * 4096), 0);
    // S_ENDPGM performs an implicit wait-idle (drains TENSORcnt) before retiring.
}

extern "C" void kernel_launch(void* const* d_in, const int* in_sizes, int n_in,
                              void* d_out, int out_size, void* d_ws, size_t ws_size,
                              hipStream_t stream)
{
    const float* gamma = (const float*)d_in[0];   // [64, 64]
    const float* noise = (const float*)d_in[1];   // [B, 64, 64]
    float* out = (float*)d_out;                   // [B, 64, 64]

    int B = in_sizes[1] / 4096;                   // 8192 matrices, one wave32 each
    sinkhorn_wmma_kernel<<<B, 32, 0, stream>>>(gamma, noise, out);
}
